// SentimentAwareBiLSTM_30837865185906
// MI455X (gfx1250) — compile-verified, matrix operand order checked by reading, not run on hardware
//
#include <hip/hip_runtime.h>
#include <hip/hip_bf16.h>

typedef __bf16 bf16_t;
typedef __attribute__((ext_vector_type(16))) __bf16 v16bf;
typedef __attribute__((ext_vector_type(2)))  __bf16 v2bf;
typedef __attribute__((ext_vector_type(8)))  float  v8f;

#define S_LEN 512
#define BATCH 64
#define IN_DIM 768
#define HID 512
#define G4H 2048

// 16-bit WMMA A/B fragment: element e of a lane-half -> k within the 32-wide k-tile.
// half 0 lanes hold k {0..7, 16..23}; half 1 lanes hold k {8..15, 24..31}.
// Note: for even e, kmap(e+1,h) == kmap(e,h)+1  (pairs are k-contiguous).
__device__ __forceinline__ int kmap(int e, int half) {
    return (e < 8) ? (e + 8 * half) : (e - 8 + 16 + 8 * half);
}
// inverse: hidden column (within a 32-wide k-tile) -> (lane-half, element)
__device__ __forceinline__ void inv_kmap(int kl, int& hk, int& e) {
  if (kl < 8)       { hk = 0; e = kl; }
  else if (kl < 16) { hk = 1; e = kl - 8; }
  else if (kl < 24) { hk = 0; e = kl - 8; }
  else              { hk = 1; e = kl - 16; }
}

// ---------------------------------------------------------------------------
// Kernel 1: convert Wi/Wh (f32 row-major [K][4H]) to bf16 in WMMA-B tile order.
// Tile (nt, kt) stored as [32 lanes][16 bf16] contiguous (32B per lane).
// ---------------------------------------------------------------------------
__global__ __launch_bounds__(256) void wconv_kernel(
    const float* __restrict__ WiF, const float* __restrict__ WiB,
    const float* __restrict__ WhF, const float* __restrict__ WhB,
    bf16_t* __restrict__ wiT, bf16_t* __restrict__ whT) {
  int gid  = blockIdx.x * 256 + threadIdx.x;
  int lane = gid & 31;
  int blk  = gid >> 5;
  const int NBLK_WI = 2 * 128 * 24;   // dirs * ntiles * ktiles (K=768)
  const int NBLK_WH = 2 * 128 * 16;   // dirs * ntiles * ktiles (K=512)
  int n_lo = lane & 15, half = lane >> 4;
  if (blk < NBLK_WI) {
    int dir = blk / (128 * 24); int r = blk % (128 * 24);
    int nt = r / 24, kt = r % 24;
    const float* src = dir ? WiB : WiF;
    bf16_t* dst = wiT + (size_t)dir * ((size_t)IN_DIM * G4H)
                      + (((size_t)(nt * 24 + kt)) * 32 + lane) * 16;
    int n = nt * 16 + n_lo;
#pragma unroll
    for (int e = 0; e < 16; ++e) {
      int k = kt * 32 + kmap(e, half);
      dst[e] = (bf16_t)src[(size_t)k * G4H + n];
    }
  } else if (blk < NBLK_WI + NBLK_WH) {
    int b2 = blk - NBLK_WI;
    int dir = b2 / (128 * 16); int r = b2 % (128 * 16);
    int nt = r / 16, kt = r % 16;
    const float* src = dir ? WhB : WhF;
    bf16_t* dst = whT + (size_t)dir * ((size_t)HID * G4H)
                      + (((size_t)(nt * 16 + kt)) * 32 + lane) * 16;
    int n = nt * 16 + n_lo;
#pragma unroll
    for (int e = 0; e < 16; ++e) {
      int k = kt * 32 + kmap(e, half);
      dst[e] = (bf16_t)src[(size_t)k * G4H + n];
    }
  }
}

// ---------------------------------------------------------------------------
// Kernel 2: x_proj = x @ Wi + bi + bh  (bf16 WMMA, f32 accum)
// M-tile = 16 batch rows at fixed t; output stored in C-fragment order:
// block ((t*4+bt)*128+nt) = [32 lanes][8 f32] (32B contiguous per lane).
// grid = (512*4 (t,bt), 4 n-chunks, 2 dirs); block = 256 (8 waves x 4 n-tiles).
// ---------------------------------------------------------------------------
__device__ __forceinline__ void load_a_tile(
    const float* __restrict__ xptr, int t, int bt, int kt,
    bf16_t (*buf)[16], int tid) {
  int idx = tid * 2;                      // 512 bf16 elements, 2 per thread
  int laneA = idx >> 4, e0 = idx & 15;    // e0 even -> k, k+1 contiguous
  int m = laneA & 15, hf = laneA >> 4;
  int k = kt * 32 + kmap(e0, hf);
  int b = bt * 16 + m;
  float2 v = *(const float2*)(xptr + ((size_t)b * S_LEN + t) * IN_DIM + k);
  v2bf pk; pk.x = (bf16_t)v.x; pk.y = (bf16_t)v.y;
  *(v2bf*)&buf[laneA][e0] = pk;
}

__global__ __launch_bounds__(256) void xproj_kernel(
    const float* __restrict__ x, const bf16_t* __restrict__ wiT,
    const float* __restrict__ biF, const float* __restrict__ bhF,
    const float* __restrict__ biB, const float* __restrict__ bhB,
    float* __restrict__ xprojF, float* __restrict__ xprojB) {
  __shared__ __align__(32) bf16_t albuf[2][32][16];   // double-buffered A tile
  int tid  = threadIdx.x;
  int lane = tid & 31, wave = tid >> 5;
  int t    = blockIdx.x >> 2;
  int bt   = blockIdx.x & 3;
  int nt0  = blockIdx.y * 32 + wave * 4;   // 4 n-tiles per wave
  int dir  = blockIdx.z;
  const bf16_t* wt = wiT + (size_t)dir * ((size_t)IN_DIM * G4H);
  const float* bi = dir ? biB : biF;
  const float* bh = dir ? bhB : bhF;
  float* xp = dir ? xprojB : xprojF;

  v8f acc[4] = {{}, {}, {}, {}};
  load_a_tile(x, t, bt, 0, albuf[0], tid);
  __syncthreads();
  for (int kt = 0; kt < 24; ++kt) {
    int cur = kt & 1;
    if (kt + 1 < 24) load_a_tile(x, t, bt, kt + 1, albuf[cur ^ 1], tid);
    v16bf a = *(const v16bf*)&albuf[cur][lane][0];
#pragma unroll
    for (int i = 0; i < 4; ++i) {
      v16bf b = *(const v16bf*)(wt + (((size_t)((nt0 + i) * 24 + kt)) * 32 + lane) * 16);
      acc[i] = __builtin_amdgcn_wmma_f32_16x16x32_bf16(
          false, a, false, b, (short)0, acc[i], false, false);
    }
    __syncthreads();
  }
#pragma unroll
  for (int i = 0; i < 4; ++i) {
    int nt = nt0 + i;
    int n  = nt * 16 + (lane & 15);
    float bias = bi[n] + bh[n];
#pragma unroll
    for (int j = 0; j < 8; ++j) acc[i][j] += bias;
    *(v8f*)(xp + ((((size_t)t * 4 + bt) * 128 + nt) * 32 + lane) * 8) = acc[i];
  }
}

// ---------------------------------------------------------------------------
// Kernel 3: sentiment-gated LSTM recurrence. 8 WGs = 2 dirs x 4 batch tiles.
// 512 threads = 16 waves; wave w owns hidden tiles {w, w+16} and computes
// their 4 gate tiles each: per k-step ONE LDS A fragment feeds 8 WMMAs.
// h double-buffered bf16 in LDS (A-fragment layout); c in VGPRs all 512 steps.
// 4 waves/SIMD -> large per-wave VGPR budget, no spill on the serial path.
// ---------------------------------------------------------------------------
__global__ __launch_bounds__(512) void lstm_rec_kernel(
    const float* __restrict__ xprojF, const float* __restrict__ xprojB,
    const bf16_t* __restrict__ whT, const float* __restrict__ sent,
    float* __restrict__ out) {
  __shared__ __align__(32) bf16_t hbuf[2][16][32][16];   // 32 KB
  int tid  = threadIdx.x;
  int lane = tid & 31, wave = tid >> 5;
  int dir  = blockIdx.x >> 2;
  int bt   = blockIdx.x & 3;
  int b0   = bt * 16;
  const float*  xp = dir ? xprojB : xprojF;
  const bf16_t* wt = whT + (size_t)dir * ((size_t)HID * G4H);
  int jt0 = wave, jt1 = wave + 16;     // two hidden n-tiles per wave
  int n = lane & 15, half = lane >> 4;
  int jg0 = jt0 * 16 + n, jg1 = jt1 * 16 + n;
  // precompute h-publish placement (inverse fragment map) for both tiles
  int kta = jg0 >> 5, hk0, he0; inv_kmap(jg0 & 31, hk0, he0);
  int ktb = jg1 >> 5, hk1, he1; inv_kmap(jg1 & 31, hk1, he1);

  // h_0 = 0
  {
    bf16_t* hz = &hbuf[0][0][0][0];
    for (int i = tid; i < 16 * 32 * 16; i += 512) hz[i] = (bf16_t)0.0f;
  }
  v8f c0 = {}, c1 = {};
  __syncthreads();

  int p = 0;
#pragma unroll 1
  for (int s = 0; s < S_LEN; ++s) {
    int t = dir ? (S_LEN - 1 - s) : s;
    // init gate accumulators from fragment-order x-projection (32B/lane each)
    size_t base = ((size_t)t * 4 + bt) * 128;
    v8f A0[4], A1[4];
#pragma unroll
    for (int gi = 0; gi < 4; ++gi) {
      A0[gi] = *(const v8f*)(xp + ((base + (gi * 32 + jt0)) * 32 + lane) * 8);
      A1[gi] = *(const v8f*)(xp + ((base + (gi * 32 + jt1)) * 32 + lane) * 8);
    }
    // gates += h_prev @ Wh : one A fragment feeds 8 gate-tile WMMAs
#pragma unroll 1
    for (int kt = 0; kt < 16; ++kt) {
      v16bf a = *(const v16bf*)&hbuf[p][kt][lane][0];
#pragma unroll
      for (int gi = 0; gi < 4; ++gi) {
        v16bf bA = *(const v16bf*)(wt + (((size_t)((gi * 32 + jt0) * 16 + kt)) * 32 + lane) * 16);
        v16bf bB = *(const v16bf*)(wt + (((size_t)((gi * 32 + jt1) * 16 + kt)) * 32 + lane) * 16);
        A0[gi] = __builtin_amdgcn_wmma_f32_16x16x32_bf16(false, a, false, bA, (short)0, A0[gi], false, false);
        A1[gi] = __builtin_amdgcn_wmma_f32_16x16x32_bf16(false, a, false, bB, (short)0, A1[gi], false, false);
      }
    }
    // nonlinearity, state update, output, publish h for next step
    int pn = p ^ 1;
#pragma unroll
    for (int j = 0; j < 8; ++j) {
      int m = j + 8 * half;
      int bb = b0 + m;
      float st = sent[(size_t)bb * S_LEN + t];   // shared by both tiles
      size_t orow = ((size_t)bb * S_LEN + t) * (2 * HID) + dir * HID;
      // tile 0
      {
        float iv = st           / (1.0f + __expf(-A0[0][j]));
        float fv = (1.0f + st)  / (1.0f + __expf(-A0[1][j]));
        float gv = tanhf(A0[2][j]);
        float ov = 1.0f / (1.0f + __expf(-A0[3][j]));
        float cn = fv * c0[j] + iv * gv;
        c0[j] = cn;
        float hv = ov * tanhf(cn);
        out[orow + jg0] = hv;
        hbuf[pn][kta][m + 16 * hk0][he0] = (bf16_t)hv;
      }
      // tile 1
      {
        float iv = st           / (1.0f + __expf(-A1[0][j]));
        float fv = (1.0f + st)  / (1.0f + __expf(-A1[1][j]));
        float gv = tanhf(A1[2][j]);
        float ov = 1.0f / (1.0f + __expf(-A1[3][j]));
        float cn = fv * c1[j] + iv * gv;
        c1[j] = cn;
        float hv = ov * tanhf(cn);
        out[orow + jg1] = hv;
        hbuf[pn][ktb][m + 16 * hk1][he1] = (bf16_t)hv;
      }
    }
    __syncthreads();
    p = pn;
  }
}

// ---------------------------------------------------------------------------
extern "C" void kernel_launch(void* const* d_in, const int* in_sizes, int n_in,
                              void* d_out, int out_size, void* d_ws, size_t ws_size,
                              hipStream_t stream) {
  const float* x    = (const float*)d_in[0];
  const float* sent = (const float*)d_in[1];
  const float* WiF  = (const float*)d_in[2];
  const float* biF  = (const float*)d_in[3];
  const float* WhF  = (const float*)d_in[4];
  const float* bhF  = (const float*)d_in[5];
  const float* WiB  = (const float*)d_in[6];
  const float* biB  = (const float*)d_in[7];
  const float* WhB  = (const float*)d_in[8];
  const float* bhB  = (const float*)d_in[9];
  float* out = (float*)d_out;

  char* ws = (char*)d_ws;
  const size_t SZ_XP = (size_t)S_LEN * BATCH * G4H * sizeof(float);   // 268.4 MB
  const size_t SZ_WI = (size_t)IN_DIM * G4H * sizeof(bf16_t);         // 3 MB
  float*  xprojF = (float*)(ws);
  float*  xprojB = (float*)(ws + SZ_XP);
  bf16_t* wiT    = (bf16_t*)(ws + 2 * SZ_XP);
  bf16_t* whT    = (bf16_t*)(ws + 2 * SZ_XP + 2 * SZ_WI);

  wconv_kernel<<<1280, 256, 0, stream>>>(WiF, WiB, WhF, WhB, wiT, whT);
  xproj_kernel<<<dim3(2048, 4, 2), 256, 0, stream>>>(
      x, wiT, biF, bhF, biB, bhB, xprojF, xprojB);
  lstm_rec_kernel<<<8, 512, 0, stream>>>(xprojF, xprojB, whT, sent, out);
}
